// APA_73658689126563
// MI455X (gfx1250) — compile-verified, matrix operand order checked by reading, not run on hardware
//
#include <hip/hip_runtime.h>

// Nearest-prototype assignment + prototype update (k-means style step) for
// MI455X (gfx1250, wave32). Scoring GEMM runs on v_wmma_f32_16x16x32_f16;
// segment sums accumulate in LDS via ds_add_f32 and flush once per block.

typedef __attribute__((ext_vector_type(16))) _Float16 v16h;
typedef __attribute__((ext_vector_type(8)))  float    v8f;

#define NPROTO 128
#define DIM    64
#define BLOCK  256
#define WAVES_PER_BLOCK (BLOCK / 32)
#define GRID   256

__device__ __forceinline__ v16h pack4(float4 f0, float4 f1, float4 f2, float4 f3) {
    v16h r;
    r[0]  = (_Float16)f0.x; r[1]  = (_Float16)f0.y; r[2]  = (_Float16)f0.z; r[3]  = (_Float16)f0.w;
    r[4]  = (_Float16)f1.x; r[5]  = (_Float16)f1.y; r[6]  = (_Float16)f1.z; r[7]  = (_Float16)f1.w;
    r[8]  = (_Float16)f2.x; r[9]  = (_Float16)f2.y; r[10] = (_Float16)f2.z; r[11] = (_Float16)f2.w;
    r[12] = (_Float16)f3.x; r[13] = (_Float16)f3.y; r[14] = (_Float16)f3.z; r[15] = (_Float16)f3.w;
    return r;
}

__device__ __forceinline__ float sq16(float4 f0, float4 f1, float4 f2, float4 f3) {
    return f0.x*f0.x + f0.y*f0.y + f0.z*f0.z + f0.w*f0.w +
           f1.x*f1.x + f1.y*f1.y + f1.z*f1.z + f1.w*f1.w +
           f2.x*f2.x + f2.y*f2.y + f2.z*f2.z + f2.w*f2.w +
           f3.x*f3.x + f3.y*f3.y + f3.z*f3.z + f3.w*f3.w;
}

__global__ void __launch_bounds__(BLOCK)
zero_ws_kernel(float* __restrict__ w, int n) {
    int i = blockIdx.x * BLOCK + threadIdx.x;
    if (i < n) w[i] = 0.0f;
}

__global__ void __launch_bounds__(BLOCK)
proto_assign_kernel(const float* __restrict__ x,
                    const float* __restrict__ protos,
                    float* __restrict__ out_assign,
                    float* __restrict__ gsums,
                    float* __restrict__ gcounts,
                    int n) {
    __shared__ float lsum[NPROTO * DIM];   // 32 KB segment-sum accumulator
    __shared__ float lcnt[NPROTO];

    const int tid   = threadIdx.x;
    const int lane  = tid & 31;
    const int wv    = tid >> 5;
    const int col   = lane & 15;          // N-column within 16x16 tile
    const int khalf = (lane >> 4) & 1;    // which K-half this lane holds
    const int kbase = khalf * 8;

    for (int i = tid; i < NPROTO * DIM; i += BLOCK) lsum[i] = 0.0f;
    if (tid < NPROTO) lcnt[tid] = 0.0f;
    __syncthreads();

    // ---- Load B (prototypes) fragments once into registers: 8 proto tiles x
    // ---- 2 K-chunks, plus exact fp32 ||p||^2 per lane's column (via shfl).
    v16h  b[8][2];
    float psq[8];
#pragma unroll
    for (int t = 0; t < 8; ++t) {
        const float* pr = protos + (t * 16 + col) * DIM;
        float s = 0.0f;
#pragma unroll
        for (int kc = 0; kc < 2; ++kc) {
            const int k0 = kbase + kc * 32;
            float4 f0 = *(const float4*)(pr + k0);
            float4 f1 = *(const float4*)(pr + k0 + 4);
            float4 f2 = *(const float4*)(pr + k0 + 16);
            float4 f3 = *(const float4*)(pr + k0 + 20);
            s += sq16(f0, f1, f2, f3);
            b[t][kc] = pack4(f0, f1, f2, f3);
        }
        s += __shfl_xor(s, 16, 32);       // add partner half's K contribution
        psq[t] = s;
    }

    const int ntiles = n >> 4;            // 16 rows per tile
    const int gw = blockIdx.x * WAVES_PER_BLOCK + wv;
    const int nw = gridDim.x * WAVES_PER_BLOCK;

    for (int tile = gw; tile < ntiles; tile += nw) {
        const int row0 = tile * 16;
        const float* xr = x + (size_t)(row0 + col) * DIM;

        // ---- A fragments (this lane's 32 of the row's 64 K values)
        v16h a[2];
#pragma unroll
        for (int kc = 0; kc < 2; ++kc) {
            const int k0 = kbase + kc * 32;
            float4 f0 = *(const float4*)(xr + k0);
            float4 f1 = *(const float4*)(xr + k0 + 4);
            float4 f2 = *(const float4*)(xr + k0 + 16);
            float4 f3 = *(const float4*)(xr + k0 + 20);
            a[kc] = pack4(f0, f1, f2, f3);
        }

        float minv[8];
        int   mini[8];
#pragma unroll
        for (int r = 0; r < 8; ++r) { minv[r] = 3.4e38f; mini[r] = 0; }

        // ---- 8 proto tiles, each = 2 chained WMMAs over K=64
#pragma unroll
        for (int t = 0; t < 8; ++t) {
            v8f c = {};
            c = __builtin_amdgcn_wmma_f32_16x16x32_f16(false, a[0], false, b[t][0],
                                                       (short)0, c, false, false);
            c = __builtin_amdgcn_wmma_f32_16x16x32_f16(false, a[1], false, b[t][1],
                                                       (short)0, c, false, false);
            const int p = t * 16 + col;
#pragma unroll
            for (int r = 0; r < 8; ++r) {
                float sc = psq[t] - 2.0f * c[r];   // ||p||^2 - 2 x.p
                if (sc < minv[r]) { minv[r] = sc; mini[r] = p; }  // t asc => first idx on tie
            }
        }

        // ---- cross-lane argmin within each 16-lane half (wave32)
#pragma unroll
        for (int mask = 1; mask <= 8; mask <<= 1) {
#pragma unroll
            for (int r = 0; r < 8; ++r) {
                float ov = __shfl_xor(minv[r], mask, 32);
                int   oi = __shfl_xor(mini[r], mask, 32);
                if (ov < minv[r] || (ov == minv[r] && oi < mini[r])) {
                    minv[r] = ov; mini[r] = oi;
                }
            }
        }
        // lanes 0-15 now hold rows 0..7 in mini[r]; lanes 16-31 hold rows 8..15

        // ---- write assignments (as float, per harness output dtype)
#pragma unroll
        for (int r = 0; r < 8; ++r) {
            if (col == r) out_assign[row0 + khalf * 8 + r] = (float)mini[r];
        }

        // ---- segment-sum into LDS: lane handles dims {2*lane, 2*lane+1}
        const float* xt = x + (size_t)row0 * DIM;
#pragma unroll
        for (int r = 0; r < 8; ++r) {
            const int pa0 = __shfl(mini[r], 0,  32);   // row r assignment
            const int pa1 = __shfl(mini[r], 16, 32);   // row 8+r assignment
            float2 v0 = *(const float2*)(xt + r * DIM + 2 * lane);
            float2 v1 = *(const float2*)(xt + (8 + r) * DIM + 2 * lane);
            atomicAdd(&lsum[pa0 * DIM + 2 * lane],     v0.x);
            atomicAdd(&lsum[pa0 * DIM + 2 * lane + 1], v0.y);
            atomicAdd(&lsum[pa1 * DIM + 2 * lane],     v1.x);
            atomicAdd(&lsum[pa1 * DIM + 2 * lane + 1], v1.y);
        }
#pragma unroll
        for (int r = 0; r < 8; ++r) {
            if (col == r) atomicAdd(&lcnt[mini[r]], 1.0f);  // 2 lanes: rows r and 8+r
        }
    }

    // ---- one flush per block to global accumulators
    __syncthreads();
    for (int i = tid; i < NPROTO * DIM; i += BLOCK) atomicAdd(&gsums[i], lsum[i]);
    if (tid < NPROTO) atomicAdd(&gcounts[tid], lcnt[tid]);
}

__global__ void __launch_bounds__(BLOCK)
finalize_kernel(const float* __restrict__ protos,
                const float* __restrict__ gsums,
                const float* __restrict__ gcounts,
                float* __restrict__ out_protos) {
    int i = blockIdx.x * BLOCK + threadIdx.x;
    if (i >= NPROTO * DIM) return;
    int p = i >> 6;                        // DIM = 64
    float c = gcounts[p];
    out_protos[i] = (c > 0.0f) ? gsums[i] / c : protos[i];
}

extern "C" void kernel_launch(void* const* d_in, const int* in_sizes, int n_in,
                              void* d_out, int out_size, void* d_ws, size_t ws_size,
                              hipStream_t stream) {
    const float* x      = (const float*)d_in[0];
    // d_in[1] = labels (unused by the reference computation)
    const float* protos = (const float*)d_in[2];
    const int n = in_sizes[0] / DIM;       // 524288

    float* gsums   = (float*)d_ws;                 // [128*64]
    float* gcounts = gsums + NPROTO * DIM;         // [128]
    float* out_assign = (float*)d_out;             // [n]
    float* out_protos = out_assign + n;            // [128*64]

    const int nacc = NPROTO * DIM + NPROTO;
    zero_ws_kernel<<<(nacc + BLOCK - 1) / BLOCK, BLOCK, 0, stream>>>(gsums, nacc);
    proto_assign_kernel<<<GRID, BLOCK, 0, stream>>>(x, protos, out_assign,
                                                    gsums, gcounts, n);
    finalize_kernel<<<(NPROTO * DIM + BLOCK - 1) / BLOCK, BLOCK, 0, stream>>>(
        protos, gsums, gcounts, out_protos);
}
